// PoolingModule_57526791962871
// MI455X (gfx1250) — compile-verified, matrix operand order checked by reading, not run on hardware
//
#include <hip/hip_runtime.h>
#include <hip/hip_bf16.h>

typedef __attribute__((ext_vector_type(16))) _Float16 v16h;
typedef __attribute__((ext_vector_type(8)))  float    v8f;

#define NGRAPH 100
#define NPG0   1000
#define N0     (NGRAPH * NPG0)    /* 100000 */
#define EPGC   10000
#define ECNT   (NGRAPH * EPGC)    /* 1000000 */
#define NF     128
#define KTOP1  500
#define KTOP2  250
#define NN1    (NGRAPH * KTOP1)   /* 50000 */
#define NN2    (NGRAPH * KTOP2)   /* 25000 */
#define LAMB_C 1.0f

// ---------------------------------------------------------------- utilities
__device__ __forceinline__ float warp_sum(float v) {
#pragma unroll
  for (int m = 16; m > 0; m >>= 1) v += __shfl_xor(v, m, 32);
  return v;
}

// ------------------------------------------------------- W pack (f32 -> f16)
// Pack 128x128 W (row-major, W[k*128+n]) into the exact per-lane WMMA B
// operand layout: Wp[((kc*8 + colTile)*32 + lane)*16 + j].
__global__ void k_pack_w(const float* __restrict__ W, _Float16* __restrict__ Wp) {
  int idx = blockIdx.x * blockDim.x + threadIdx.x;
  if (idx >= 4 * 8 * 32 * 16) return;
  int j    = idx & 15;
  int lane = (idx >> 4) & 31;
  int ct   = (idx >> 9) & 7;
  int kc   = idx >> 12;
  int ncol = (lane & 15) + ct * 16;
  int offB = (lane < 16) ? 0 : 8;
  int k    = kc * 32 + ((j < 8) ? (offB + j) : (16 + offB + (j - 8)));
  Wp[idx] = (_Float16)W[k * NF + ncol];
}

// ------------------------------------------------------------- WMMA GEMM
// H[n,128] = X[n,128] @ W[128,128]; one 16x16 tile per wave, 8 waves/block
// covering the 128 output columns; K=128 via 4x v_wmma_f32_16x16x32_f16.
__global__ void __launch_bounds__(256)
k_gemm(const float* __restrict__ X, const _Float16* __restrict__ Wp,
       float* __restrict__ Hout, int n) {
  int tileRow = blockIdx.x;
  int wave = threadIdx.x >> 5;       // column tile 0..7
  int lane = threadIdx.x & 31;
  int mrow = lane & 15;
  int off  = (lane < 16) ? 0 : 8;    // K sub-offset per A-layout
  int grow = tileRow * 16 + mrow;
  int rowc = (grow < n) ? grow : (n - 1);   // clamp; stores guarded below
  const float* xrow = X + (size_t)rowc * NF;
  v8f c = {};
#pragma unroll
  for (int kc = 0; kc < 4; ++kc) {
    int k0 = kc * 32;
    const float4* pp = (const float4*)(xrow + k0 + off);
    float4 p0 = pp[0], p1 = pp[1];
    const float4* qq = (const float4*)(xrow + k0 + 16 + off);
    float4 q0 = qq[0], q1 = qq[1];
    v16h a;
    a[0]  = (_Float16)p0.x; a[1]  = (_Float16)p0.y;
    a[2]  = (_Float16)p0.z; a[3]  = (_Float16)p0.w;
    a[4]  = (_Float16)p1.x; a[5]  = (_Float16)p1.y;
    a[6]  = (_Float16)p1.z; a[7]  = (_Float16)p1.w;
    a[8]  = (_Float16)q0.x; a[9]  = (_Float16)q0.y;
    a[10] = (_Float16)q0.z; a[11] = (_Float16)q0.w;
    a[12] = (_Float16)q1.x; a[13] = (_Float16)q1.y;
    a[14] = (_Float16)q1.z; a[15] = (_Float16)q1.w;
    v16h b = *(const v16h*)(Wp + (((kc * 8 + wave) * 32 + lane) * 16));
    c = __builtin_amdgcn_wmma_f32_16x16x32_f16(
        false, a, false, b, (short)0, c, false, false);
  }
  int colBase = wave * 16 + (lane & 15);
  int rbase = (lane < 16) ? 0 : 8;   // D layout: VGPR r -> M=r / M=8+r
#pragma unroll
  for (int r = 0; r < 8; ++r) {
    int row = tileRow * 16 + rbase + r;
    if (row < n) Hout[(size_t)row * NF + colBase] = c[r];
  }
}

// ------------------------------------------------------------- GCN pieces
__global__ void k_fill_ones(float* __restrict__ w, int n) {
  int i = blockIdx.x * blockDim.x + threadIdx.x;
  if (i < n) w[i] = 1.0f;
}

__global__ void k_deg(const int* __restrict__ dst, const float* __restrict__ w,
                      float* __restrict__ deg, int ne) {
  int e = blockIdx.x * blockDim.x + threadIdx.x;
  if (e < ne) atomicAdd(deg + dst[e], w[e]);
}

__global__ void k_dinv(float* __restrict__ deg, int n) {
  int i = blockIdx.x * blockDim.x + threadIdx.x;
  if (i >= n) return;
  float dg = deg[i] + 1.0f;                       // self loop
  deg[i] = rsqrtf(fmaxf(dg, 1e-12f));             // deg := dinv in place
}

// one wave per edge: AGG[dst] += H[src] * (dinv[src]*w*dinv[dst])
__global__ void __launch_bounds__(256)
k_edge_agg(const int* __restrict__ src, const int* __restrict__ dst,
           const float* __restrict__ w, const float* __restrict__ dinv,
           const float* __restrict__ H, float* __restrict__ AGG, int ne) {
  int wid = (blockIdx.x * blockDim.x + threadIdx.x) >> 5;
  int lane = threadIdx.x & 31;
  if (wid >= ne) return;
  int s = src[wid], d = dst[wid];
  float coef = dinv[s] * w[wid] * dinv[d];
  float4 hv = *(const float4*)(H + (size_t)s * NF + lane * 4);
  float* o = AGG + (size_t)d * NF + lane * 4;
  atomicAdd(o + 0, hv.x * coef);
  atomicAdd(o + 1, hv.y * coef);
  atomicAdd(o + 2, hv.z * coef);
  atomicAdd(o + 3, hv.w * coef);
}

// Xout := relu(Xout + H * dinv^2 + b)   (in place on the AGG buffer)
__global__ void k_combine(float* __restrict__ Xout, const float* __restrict__ H,
                          const float* __restrict__ dinv,
                          const float* __restrict__ bias, int n) {
  int idx = blockIdx.x * blockDim.x + threadIdx.x;
  if (idx >= n * NF) return;
  int i = idx >> 7, f = idx & 127;
  float di = dinv[i];
  float v = Xout[idx] + H[idx] * di * di + bias[f];
  Xout[idx] = fmaxf(v, 0.0f);
}

// --------------------------------------------------------- info score
__global__ void __launch_bounds__(256)
k_mean_agg(const int* __restrict__ src, const int* __restrict__ dst,
           const float* __restrict__ w, const float* __restrict__ X,
           float* __restrict__ MS, float* __restrict__ deg2, int ne) {
  int wid = (blockIdx.x * blockDim.x + threadIdx.x) >> 5;
  int lane = threadIdx.x & 31;
  if (wid >= ne) return;
  int s = src[wid], d = dst[wid];
  float wt = w[wid];
  float4 xv = *(const float4*)(X + (size_t)s * NF + lane * 4);
  float* o = MS + (size_t)d * NF + lane * 4;
  atomicAdd(o + 0, xv.x * wt);
  atomicAdd(o + 1, xv.y * wt);
  atomicAdd(o + 2, xv.z * wt);
  atomicAdd(o + 3, xv.w * wt);
  if (lane == 0) atomicAdd(deg2 + d, wt);
}

__global__ void __launch_bounds__(256)
k_score(const float* __restrict__ X, const float* __restrict__ MS,
        const float* __restrict__ deg2, float* __restrict__ score, int n) {
  int wid = (blockIdx.x * blockDim.x + threadIdx.x) >> 5;
  int lane = threadIdx.x & 31;
  if (wid >= n) return;
  float dg = fmaxf(deg2[wid], 1e-12f);
  float4 xv = *(const float4*)(X + (size_t)wid * NF + lane * 4);
  float4 mv = *(const float4*)(MS + (size_t)wid * NF + lane * 4);
  float acc = fabsf(xv.x - mv.x / dg) + fabsf(xv.y - mv.y / dg) +
              fabsf(xv.z - mv.z / dg) + fabsf(xv.w - mv.w / dg);
  acc = warp_sum(acc);
  if (lane == 0) score[wid] = acc;
}

// --------------------------------------------------------- per-graph top-k
__global__ void __launch_bounds__(256)
k_topk(const float* __restrict__ score, int* __restrict__ perm,
       int* __restrict__ inv, int npg, int kk) {
  __shared__ float sc[1024];
  int b = blockIdx.x;
  for (int i = threadIdx.x; i < npg; i += blockDim.x)
    sc[i] = score[b * npg + i];
  __syncthreads();
  for (int i = threadIdx.x; i < npg; i += blockDim.x) {
    float si = sc[i];
    int rank = 0;
    for (int j = 0; j < npg; ++j) {
      float sj = sc[j];
      rank += (sj > si) || (sj == si && j < i);
    }
    if (rank < kk) {
      int nid = b * npg + i;
      int nw  = b * kk + rank;
      perm[nw] = nid;
      inv[nid] = nw;
    }
  }
}

__global__ void k_gather(const float* __restrict__ X, const int* __restrict__ perm,
                         float* __restrict__ Xk, int new_n) {
  int idx = blockIdx.x * blockDim.x + threadIdx.x;
  if (idx >= new_n * NF) return;
  int i = idx >> 7, f = idx & 127;
  Xk[idx] = X[(size_t)perm[i] * NF + f];
}

__global__ void k_remap(const int* __restrict__ src, const int* __restrict__ dst,
                        const float* __restrict__ w, const int* __restrict__ inv,
                        int* __restrict__ ns, int* __restrict__ nd,
                        float* __restrict__ wk, unsigned char* __restrict__ val,
                        int ne) {
  int e = blockIdx.x * blockDim.x + threadIdx.x;
  if (e >= ne) return;
  int s = inv[src[e]], d = inv[dst[e]];
  bool v = (s >= 0) && (d >= 0);
  ns[e] = v ? s : 0;
  nd[e] = v ? d : 0;
  wk[e] = v ? w[e] : 0.0f;
  val[e] = v ? 1 : 0;
}

// per-node attention dots: s1 = x . att[0:128], s2 = x . att[128:256]
__global__ void __launch_bounds__(256)
k_attdot(const float* __restrict__ Xk, const float* __restrict__ att,
         float* __restrict__ s1, float* __restrict__ s2, int n) {
  int wid = (blockIdx.x * blockDim.x + threadIdx.x) >> 5;
  int lane = threadIdx.x & 31;
  if (wid >= n) return;
  float4 xv = *(const float4*)(Xk + (size_t)wid * NF + lane * 4);
  float4 al = *(const float4*)(att + lane * 4);
  float4 ah = *(const float4*)(att + NF + lane * 4);
  float a1 = xv.x * al.x + xv.y * al.y + xv.z * al.z + xv.w * al.w;
  float a2 = xv.x * ah.x + xv.y * ah.y + xv.z * ah.z + xv.w * ah.w;
  a1 = warp_sum(a1);
  a2 = warp_sum(a2);
  if (lane == 0) { s1[wid] = a1; s2[wid] = a2; }
}

// valid logits are >= 0 (relu + lamb*w, w >= 0) so int-punned atomicMax vs
// zero-initialized mx reproduces segment_max with the -inf -> 0 fixup.
__global__ void k_logit_max(const int* __restrict__ ns, const int* __restrict__ nd,
                            const float* __restrict__ wk,
                            const unsigned char* __restrict__ val,
                            const float* __restrict__ s1, const float* __restrict__ s2,
                            float* __restrict__ mx, int ne) {
  int e = blockIdx.x * blockDim.x + threadIdx.x;
  if (e >= ne || !val[e]) return;
  float lg = fmaxf(s1[ns[e]] + s2[nd[e]], 0.0f) + LAMB_C * wk[e];
  atomicMax((int*)(mx + nd[e]), __float_as_int(lg));
}

__global__ void k_exp(const int* __restrict__ ns, const int* __restrict__ nd,
                      const float* __restrict__ wk,
                      const unsigned char* __restrict__ val,
                      const float* __restrict__ s1, const float* __restrict__ s2,
                      const float* __restrict__ mx, float* __restrict__ ex,
                      float* __restrict__ den, int ne) {
  int e = blockIdx.x * blockDim.x + threadIdx.x;
  if (e >= ne) return;
  float r = 0.0f;
  if (val[e]) {
    float lg = fmaxf(s1[ns[e]] + s2[nd[e]], 0.0f) + LAMB_C * wk[e];
    r = __expf(lg - mx[nd[e]]);
    atomicAdd(den + nd[e], r);
  }
  ex[e] = r;
}

__global__ void k_norm(const int* __restrict__ nd, const float* __restrict__ ex,
                       const float* __restrict__ den, float* __restrict__ wout,
                       int ne) {
  int e = blockIdx.x * blockDim.x + threadIdx.x;
  if (e >= ne) return;
  wout[e] = ex[e] / fmaxf(den[nd[e]], 1e-12f);
}

// --------------------------------------------------------- readouts
__global__ void k_readout(const float* __restrict__ X, int kk,
                          float* __restrict__ R) {
  int idx = blockIdx.x * blockDim.x + threadIdx.x;
  if (idx >= NGRAPH * NF) return;
  int b = idx >> 7, f = idx & 127;
  float mx = -3.4e38f, sm = 0.0f;
  const float* base = X + ((size_t)b * kk) * NF + f;
  for (int j = 0; j < kk; ++j) {
    float v = base[(size_t)j * NF];
    mx = fmaxf(mx, v);
    sm += v;
  }
  R[b * 2 * NF + f] = mx;
  R[b * 2 * NF + NF + f] = sm / (float)kk;
}

__global__ void k_final_readout(const float* __restrict__ R1,
                                const float* __restrict__ R2,
                                const float* __restrict__ R3,
                                float* __restrict__ out, int n) {
  int i = blockIdx.x * blockDim.x + threadIdx.x;
  if (i >= n) return;
  out[i] = fmaxf(R1[i], 0.0f) + fmaxf(R2[i], 0.0f) + fmaxf(R3[i], 0.0f);
}

// --------------------------------------------------------- output writers
__global__ void k_copyf(const float* __restrict__ a, float* __restrict__ o, int n) {
  int i = blockIdx.x * blockDim.x + threadIdx.x;
  if (i < n) o[i] = a[i];
}
__global__ void k_out_edges(const int* __restrict__ s, const int* __restrict__ d,
                            const float* __restrict__ w, int* __restrict__ osrc,
                            int* __restrict__ odst, float* __restrict__ ow, int ne) {
  int e = blockIdx.x * blockDim.x + threadIdx.x;
  if (e >= ne) return;
  osrc[e] = s[e];
  odst[e] = d[e];
  ow[e] = w[e];
}
__global__ void k_batchout(int* __restrict__ o, int kk, int n) {
  int i = blockIdx.x * blockDim.x + threadIdx.x;
  if (i < n) o[i] = i / kk;
}

// =========================================================== host driver
extern "C" void kernel_launch(void* const* d_in, const int* in_sizes, int n_in,
                              void* d_out, int out_size, void* d_ws, size_t ws_size,
                              hipStream_t stream) {
  (void)in_sizes; (void)n_in; (void)out_size; (void)ws_size;
  const float* x0   = (const float*)d_in[0];
  const int*   ei   = (const int*)d_in[1];
  const float* W1   = (const float*)d_in[3];
  const float* b1   = (const float*)d_in[4];
  const float* W2   = (const float*)d_in[5];
  const float* b2   = (const float*)d_in[6];
  const float* W3   = (const float*)d_in[7];
  const float* b3   = (const float*)d_in[8];
  const float* att1 = (const float*)d_in[9];
  const float* att2 = (const float*)d_in[10];
  const int* src0 = ei;
  const int* dst0 = ei + ECNT;

  char* p = (char*)d_ws;
  auto carve = [&](size_t bytes) -> void* {
    void* r = (void*)p;
    p += (bytes + 255) & ~(size_t)255;
    return r;
  };
  float*    XA   = (float*)carve((size_t)N0 * NF * 4);
  float*    XB   = (float*)carve((size_t)N0 * NF * 4);
  float*    H    = (float*)carve((size_t)N0 * NF * 4);   // GEMM out / mean buf
  _Float16* WP1  = (_Float16*)carve(16384 * 2);
  _Float16* WP2  = (_Float16*)carve(16384 * 2);
  _Float16* WP3  = (_Float16*)carve(16384 * 2);
  float*    wOne = (float*)carve((size_t)ECNT * 4);
  float*    WB1  = (float*)carve((size_t)ECNT * 4);
  float*    WB2  = (float*)carve((size_t)ECNT * 4);
  float*    EXB  = (float*)carve((size_t)ECNT * 4);
  int*      ES1  = (int*)carve((size_t)ECNT * 4);
  int*      ED1  = (int*)carve((size_t)ECNT * 4);
  int*      ES2  = (int*)carve((size_t)ECNT * 4);
  int*      ED2  = (int*)carve((size_t)ECNT * 4);
  unsigned char* VAL = (unsigned char*)carve(ECNT);
  float*    deg  = (float*)carve((size_t)N0 * 4);
  float*    deg2 = (float*)carve((size_t)N0 * 4);
  float*    scr  = (float*)carve((size_t)N0 * 4);
  int*      inv  = (int*)carve((size_t)N0 * 4);
  int*      perm = (int*)carve((size_t)NN1 * 4);
  float*    s1b  = (float*)carve((size_t)NN1 * 4);
  float*    s2b  = (float*)carve((size_t)NN1 * 4);
  float*    mxb  = (float*)carve((size_t)NN1 * 4);
  float*    denb = (float*)carve((size_t)NN1 * 4);
  float*    R1   = (float*)carve((size_t)NGRAPH * 256 * 4);
  float*    R2   = (float*)carve((size_t)NGRAPH * 256 * 4);
  float*    R3   = (float*)carve((size_t)NGRAPH * 256 * 4);

  const int TB = 256;
  const int EBLK  = (ECNT + TB - 1) / TB;   // thread-per-edge grids
  const int EWAVE = (ECNT + 7) / 8;         // wave-per-edge grids

  // pack weights once per launch
  k_pack_w<<<64, TB, 0, stream>>>(W1, WP1);
  k_pack_w<<<64, TB, 0, stream>>>(W2, WP2);
  k_pack_w<<<64, TB, 0, stream>>>(W3, WP3);
  k_fill_ones<<<EBLK, TB, 0, stream>>>(wOne, ECNT);

  auto run_gcn = [&](const float* Xin, const int* es, const int* ed,
                     const float* wv, const _Float16* Wp, const float* bias,
                     float* Xout, int n) {
    hipMemsetAsync(Xout, 0, (size_t)n * NF * 4, stream);
    hipMemsetAsync(deg, 0, (size_t)n * 4, stream);
    k_gemm<<<(n + 15) / 16, TB, 0, stream>>>(Xin, Wp, H, n);
    k_deg<<<EBLK, TB, 0, stream>>>(ed, wv, deg, ECNT);
    k_dinv<<<(n + TB - 1) / TB, TB, 0, stream>>>(deg, n);
    k_edge_agg<<<EWAVE, TB, 0, stream>>>(es, ed, wv, deg, H, Xout, ECNT);
    k_combine<<<(n * NF + TB - 1) / TB, TB, 0, stream>>>(Xout, H, deg, bias, n);
  };

  auto run_pool = [&](const float* Xin, const int* es, const int* ed,
                      const float* wv, const float* att, int npg, int kk,
                      float* Xk, int* eso, int* edo, float* wo, float* Rout,
                      int n, int new_n) {
    hipMemsetAsync(H, 0, (size_t)n * NF * 4, stream);
    hipMemsetAsync(deg2, 0, (size_t)n * 4, stream);
    k_mean_agg<<<EWAVE, TB, 0, stream>>>(es, ed, wv, Xin, H, deg2, ECNT);
    k_score<<<(n + 7) / 8, TB, 0, stream>>>(Xin, H, deg2, scr, n);
    hipMemsetAsync(inv, 0xFF, (size_t)n * 4, stream);   // -1
    k_topk<<<NGRAPH, TB, 0, stream>>>(scr, perm, inv, npg, kk);
    k_gather<<<(new_n * NF + TB - 1) / TB, TB, 0, stream>>>(Xin, perm, Xk, new_n);
    k_remap<<<EBLK, TB, 0, stream>>>(es, ed, wv, inv, eso, edo, wo, VAL, ECNT);
    k_attdot<<<(new_n + 7) / 8, TB, 0, stream>>>(Xk, att, s1b, s2b, new_n);
    hipMemsetAsync(mxb, 0, (size_t)new_n * 4, stream);
    hipMemsetAsync(denb, 0, (size_t)new_n * 4, stream);
    k_logit_max<<<EBLK, TB, 0, stream>>>(eso, edo, wo, VAL, s1b, s2b, mxb, ECNT);
    k_exp<<<EBLK, TB, 0, stream>>>(eso, edo, wo, VAL, s1b, s2b, mxb, EXB, denb, ECNT);
    k_norm<<<EBLK, TB, 0, stream>>>(edo, EXB, denb, wo, ECNT);
    k_readout<<<(NGRAPH * NF + TB - 1) / TB, TB, 0, stream>>>(Xk, kk, Rout);
  };

  // ---- layer 1
  run_gcn(x0, src0, dst0, wOne, WP1, b1, XA, N0);
  run_pool(XA, src0, dst0, wOne, att1, NPG0, KTOP1, XB, ES1, ED1, WB1, R1,
           N0, NN1);
  // ---- layer 2
  run_gcn(XB, ES1, ED1, WB1, WP2, b2, XA, NN1);
  run_pool(XA, ES1, ED1, WB1, att2, KTOP1, KTOP2, XB, ES2, ED2, WB2, R2,
           NN1, NN2);
  // ---- layer 3
  run_gcn(XB, ES2, ED2, WB2, WP3, b3, XA, NN2);
  k_readout<<<(NGRAPH * NF + TB - 1) / TB, TB, 0, stream>>>(XA, KTOP2, R3);

  // ---- outputs: [xf | edge_index(2E) | w(E) | batch(NN2) | readouts]
  float* fo = (float*)d_out;
  int*   io = (int*)d_out;
  const int XF_N  = NN2 * NF;            // 3,200,000
  const int EI_O  = XF_N;                // src at EI_O, dst at EI_O+E
  const int W_O   = EI_O + 2 * ECNT;     // 5,200,000
  const int B_O   = W_O + ECNT;          // 6,200,000
  const int R_O   = B_O + NN2;           // 6,225,000

  k_copyf<<<(XF_N + TB - 1) / TB, TB, 0, stream>>>(XA, fo, XF_N);
  k_out_edges<<<EBLK, TB, 0, stream>>>(ES2, ED2, WB2, io + EI_O,
                                       io + EI_O + ECNT, fo + W_O, ECNT);
  k_batchout<<<(NN2 + TB - 1) / TB, TB, 0, stream>>>(io + B_O, KTOP2, NN2);
  k_final_readout<<<(NGRAPH * 256 + TB - 1) / TB, TB, 0, stream>>>(
      R1, R2, R3, fo + R_O, NGRAPH * 256);
}
